// Code_Agent_17231408791671
// MI455X (gfx1250) — matrix operand order repaired and checked by reading, not compiled, not run on hardware
//
#include <hip/hip_runtime.h>
#include <hip/hip_bf16.h>

// ---------------------------------------------------------------------------
// MI455X (gfx1250, wave32) fused GRU + intent + attention network.
// All GEMMs via v_wmma_f32_16x16x32_bf16; weights pre-packed into WMMA-B
// fragment order in workspace so each lane loads 32 contiguous bytes.
// ---------------------------------------------------------------------------

typedef __attribute__((ext_vector_type(16))) __bf16 v16bf;
typedef __attribute__((ext_vector_type(8)))  float  v8f;
typedef unsigned int uint;

// ---- constants of the problem -------------------------------------------
#define ROWS   32768          // B*N
#define HDIM   256
#define INDIM  256
#define NHDIM  256
#define NADIM  32
#define IDIM   64
#define ADIM   64
#define NAGENT 8

// ---------------------------------------------------------------------------
// Fragment loaders (CDNA5 wave32 layouts, CDNA5 ISA §7.12.2)
// ---------------------------------------------------------------------------

// A: 16x32 bf16, row-major in LDS with row stride ldk (elements).
// lane L (m = L&15, g = L>>4); half j -> K = (j&7) + g*8 + (j>>3)*16.
// Loaded as 8 dword pairs (consecutive halves are K-adjacent).
__device__ __forceinline__ v16bf load_a_frag(const __bf16* A, int ldk, int k0) {
    const int lane = threadIdx.x & 31;
    const int m = lane & 15, g = lane >> 4;
    const __bf16* p = A + m * ldk + k0;
    v16bf a;
    uint* ap = reinterpret_cast<uint*>(&a);
#pragma unroll
    for (int jj = 0; jj < 8; ++jj) {
        int k = (jj & 3) * 2 + g * 8 + (jj >> 2) * 16;
        ap[jj] = *reinterpret_cast<const uint*>(p + k);   // ds_load_b32
    }
    return a;
}

// B: 32x16 bf16 fragment, pre-packed in global memory: frag f holds 512
// halves, lane L's 16 halves contiguous at f*512 + L*16 (32B aligned).
__device__ __forceinline__ v16bf load_b_frag(const __bf16* Wp, int frag) {
    const int lane = threadIdx.x & 31;
    return *reinterpret_cast<const v16bf*>(Wp + (size_t)frag * 512 + lane * 16);
}

// One 16(M)x16(N) f32 tile, K-loop in steps of 32.
__device__ __forceinline__ v8f gemm_tile(const __bf16* A, int ldk, int K,
                                         const __bf16* Wp, int ntile) {
    v8f c = {0.f, 0.f, 0.f, 0.f, 0.f, 0.f, 0.f, 0.f};
    const int kt = K >> 5;
    for (int t = 0; t < kt; ++t) {
        v16bf a = load_a_frag(A, ldk, t * 32);
        v16bf b = load_b_frag(Wp, ntile * kt + t);
        c = __builtin_amdgcn_wmma_f32_16x16x32_bf16(
                false, a, false, b, (short)0, c, false, false);
    }
    return c;
}

// D (16x16 f32): lane L holds col n=L&15, rows i + 8*(L>>4).
__device__ __forceinline__ void store_tile(float* D, int ldn, v8f c) {
    const int lane = threadIdx.x & 31;
    const int n = lane & 15, hi = lane >> 4;
#pragma unroll
    for (int i = 0; i < 8; ++i) D[(i + 8 * hi) * ldn + n] = c[i];
}

__device__ __forceinline__ void store_tile_bias(float* D, int ldn, v8f c,
                                                const float* bias, int n0) {
    const int lane = threadIdx.x & 31;
    const int n = lane & 15, hi = lane >> 4;
    const float b = bias[n0 + n];
#pragma unroll
    for (int i = 0; i < 8; ++i) D[(i + 8 * hi) * ldn + n] = c[i] + b;
}

// ---------------------------------------------------------------------------
// Weight repack: f32 row-major (N x K) -> bf16 WMMA-B fragments.
// frag f = nt*(K/32)+kt ; within frag: [lane][j], K = (j&7) + (L>>4)*8 + (j>>3)*16.
// ---------------------------------------------------------------------------
__global__ void pack_b_kernel(const float* __restrict__ W, __bf16* __restrict__ out,
                              int N, int K) {
    int idx = blockIdx.x * blockDim.x + threadIdx.x;
    if (idx >= N * K) return;
    int j  = idx & 15;
    int L  = (idx >> 4) & 31;
    int f  = idx >> 9;
    int kt = f % (K >> 5);
    int nt = f / (K >> 5);
    int n  = nt * 16 + (L & 15);
    int k  = kt * 32 + (j & 7) + (L >> 4) * 8 + (j >> 3) * 16;
    out[idx] = (__bf16)W[(size_t)n * K + k];
}

__global__ void zero_kernel(float* p, int n) {
    int i = blockIdx.x * blockDim.x + threadIdx.x;
    if (i < n) p[i] = 0.f;
}

// ---------------------------------------------------------------------------
// Kernel A: fc1 -> GRU -> h ; [h|la] -> in1 -> z (+BN partial sums)
// 16 rows per 256-thread (8-wave) workgroup.
// ---------------------------------------------------------------------------
__global__ void gru_in1_kernel(
    const float* __restrict__ inputs, const float* __restrict__ la,
    const float* __restrict__ hin,
    const float* __restrict__ fc1_b, const float* __restrict__ gbih,
    const float* __restrict__ gbhh, const float* __restrict__ in1_b,
    const __bf16* __restrict__ Wfc1, const __bf16* __restrict__ Wih,
    const __bf16* __restrict__ Whh, const __bf16* __restrict__ Win1,
    float* __restrict__ h_out, float* __restrict__ z_out,
    float* __restrict__ bn_sum, float* __restrict__ bn_sumsq) {

    extern __shared__ char smem[];
    __bf16* inA = (__bf16*)smem;                    // 16x256
    __bf16* xbf = inA + 16 * 256;                   // 16x256
    __bf16* hbf = xbf + 16 * 256;                   // 16x256
    __bf16* hla = hbf + 16 * 256;                   // 16x288
    float*  gi  = (float*)(hla + 16 * 288);         // 16x768
    float*  gh  = gi + 16 * 768;                    // 16x768
    float*  zt  = gh + 16 * 768;                    // 16x256

    const int tid = threadIdx.x;
    const int wv  = tid >> 5;
    const int rowBase = blockIdx.x * 16;

    // stage inputs + h as bf16
    for (int idx = tid; idx < 16 * 256; idx += 256) {
        int r = idx >> 8, c = idx & 255;
        size_t g = (size_t)(rowBase + r) * 256 + c;
        inA[idx] = (__bf16)inputs[g];
        hbf[idx] = (__bf16)hin[g];
    }
    __syncthreads();

    // fc1: 16 n-tiles -> gi[:, :256] (temp)
    for (int nt = wv; nt < 16; nt += 8)
        store_tile(gi + nt * 16, 256, gemm_tile(inA, 256, 256, Wfc1, nt));
    __syncthreads();

    // relu(fc1 + b) -> x (bf16)
    for (int idx = tid; idx < 16 * 256; idx += 256) {
        float v = gi[idx] + fc1_b[idx & 255];
        xbf[idx] = (__bf16)(v > 0.f ? v : 0.f);
    }
    __syncthreads();

    // GRU mats: gi = x @ Wih^T, gh = h @ Whh^T  (48 + 48 tiles)
    for (int t = wv; t < 96; t += 8) {
        if (t < 48)
            store_tile(gi + t * 16, 768, gemm_tile(xbf, 256, 256, Wih, t));
        else {
            int nt = t - 48;
            store_tile(gh + nt * 16, 768, gemm_tile(hbf, 256, 256, Whh, nt));
        }
    }
    __syncthreads();

    // GRU pointwise -> h, and [h | la] staging
    for (int idx = tid; idx < 16 * 256; idx += 256) {
        int r = idx >> 8, c = idx & 255;
        float ir = gi[r * 768 + c]       + gbih[c];
        float iz = gi[r * 768 + 256 + c] + gbih[256 + c];
        float in_ = gi[r * 768 + 512 + c] + gbih[512 + c];
        float hr = gh[r * 768 + c]       + gbhh[c];
        float hz = gh[r * 768 + 256 + c] + gbhh[256 + c];
        float hn = gh[r * 768 + 512 + c] + gbhh[512 + c];
        float rg = 1.f / (1.f + __expf(-(ir + hr)));
        float zg = 1.f / (1.f + __expf(-(iz + hz)));
        float ng = tanhf(in_ + rg * hn);
        float hp = hin[(size_t)(rowBase + r) * 256 + c];
        float h  = (1.f - zg) * ng + zg * hp;
        h_out[(size_t)(rowBase + r) * 256 + c] = h;
        hla[r * 288 + c] = (__bf16)h;
    }
    for (int idx = tid; idx < 16 * 32; idx += 256) {
        int r = idx >> 5, c = idx & 31;
        hla[r * 288 + 256 + c] = (__bf16)la[(size_t)(rowBase + r) * 32 + c];
    }
    __syncthreads();

    // in1: [h|la](K=288) @ Win1^T -> z (16 n-tiles)
    for (int nt = wv; nt < 16; nt += 8)
        store_tile(zt + nt * 16, 256, gemm_tile(hla, 288, 288, Win1, nt));
    __syncthreads();

    // write z + BN column partial sums
    for (int idx = tid; idx < 16 * 256; idx += 256) {
        int r = idx >> 8, c = idx & 255;
        z_out[(size_t)(rowBase + r) * 256 + c] = zt[idx] + in1_b[c];
    }
    {
        float s = 0.f, s2 = 0.f;
        const float bc = in1_b[tid];
#pragma unroll 4
        for (int r = 0; r < 16; ++r) {
            float z = zt[r * 256 + tid] + bc;
            s += z; s2 += z * z;
        }
        atomicAdd(&bn_sum[tid], s);
        atomicAdd(&bn_sumsq[tid], s2);
    }
}

// ---------------------------------------------------------------------------
// BN finalize: fold train-mode batchnorm into affine z*s + t.
// ---------------------------------------------------------------------------
__global__ void bn_finalize_kernel(const float* __restrict__ sum,
                                   const float* __restrict__ sumsq,
                                   const float* __restrict__ g,
                                   const float* __restrict__ b,
                                   float* __restrict__ S, float* __restrict__ T) {
    int c = threadIdx.x;
    float mu  = sum[c] * (1.f / 32768.f);
    float var = sumsq[c] * (1.f / 32768.f) - mu * mu;
    float s   = g[c] * rsqrtf(var + 1e-5f);
    S[c] = s;
    T[c] = b[c] - mu * s;
}

// ---------------------------------------------------------------------------
// Kernel C: BN+leaky -> in2 -> intent -> q,k,v -> 8x8 attention -> fc2.
// 64 rows (8 agent-batches) per 256-thread workgroup.
// ---------------------------------------------------------------------------
__global__ void attn_head_kernel(
    const float* __restrict__ z_in, const float* __restrict__ bnS,
    const float* __restrict__ bnB, const float* __restrict__ in2_b,
    const float* __restrict__ eps, const float* __restrict__ wq_b,
    const float* __restrict__ wk_b, const float* __restrict__ wv_b,
    const float* __restrict__ fc2_b,
    const __bf16* __restrict__ Win2, const __bf16* __restrict__ Wq,
    const __bf16* __restrict__ Wk,  const __bf16* __restrict__ Wv,
    const __bf16* __restrict__ Wfc2,
    const float* __restrict__ h_in,
    float* __restrict__ localQ, float* __restrict__ intent_out,
    float* __restrict__ iemb_out) {

    extern __shared__ char smem[];
    __bf16* act = (__bf16*)smem;          // 64x256 (BN+leaky activations)
    __bf16* ihc = act + 64 * 256;         // 64x320 [intent | h]
    __bf16* hc  = ihc + 64 * 320;         // 64x320 [h | combined]
    float*  ipf = (float*)(hc + 64 * 320);// 64x128
    float*  qf  = ipf + 64 * 128;         // 64x64
    float*  kf  = qf + 64 * 64;           // 64x64
    float*  vf  = kf + 64 * 64;           // 64x64

    const int tid = threadIdx.x;
    const int wv  = tid >> 5;
    const int rowBase = blockIdx.x * 64;

    // BN affine + leaky relu; stage h into both concat buffers
    for (int idx = tid; idx < 64 * 256; idx += 256) {
        int r = idx >> 8, c = idx & 255;
        float a = z_in[(size_t)(rowBase + r) * 256 + c] * bnS[c] + bnB[c];
        act[idx] = (__bf16)(a > 0.f ? a : 0.01f * a);
        float hv = h_in[(size_t)(rowBase + r) * 256 + c];
        ihc[r * 320 + 64 + c] = (__bf16)hv;
        hc[r * 320 + c]       = (__bf16)hv;
    }
    __syncthreads();

    // in2: act(K=256) @ Win2^T -> ip (4 m-tiles x 8 n-tiles)
    for (int t = wv; t < 32; t += 8) {
        int mt = t >> 3, nt = t & 7;
        store_tile(ipf + mt * 16 * 128 + nt * 16, 128,
                   gemm_tile(act + mt * 16 * 256, 256, 256, Win2, nt));
    }
    __syncthreads();

    // intent_embed = ip + b2 (written twice is same value per reference)
    for (int idx = tid; idx < 64 * 128; idx += 256) {
        int r = idx >> 7, c = idx & 127;
        float v = ipf[idx] + in2_b[c];
        ipf[idx] = v;
        iemb_out[(size_t)(rowBase + r) * 128 + c] = v;
    }
    __syncthreads();
    // intent = mean + eps * max(exp(logstd), floor)
    for (int idx = tid; idx < 64 * 64; idx += 256) {
        int r = idx >> 6, a = idx & 63;
        float mean = ipf[r * 128 + a];
        float sd   = fmaxf(__expf(ipf[r * 128 + 64 + a]), 0.002f);
        float it   = mean + eps[(size_t)(rowBase + r) * 64 + a] * sd;
        intent_out[(size_t)(rowBase + r) * 64 + a] = it;
        ihc[r * 320 + a] = (__bf16)it;
    }
    __syncthreads();

    // q = intent@Wq^T, k = intent@Wk^T, v = [intent|h]@Wv^T  (48 tiles)
    for (int t = wv; t < 48; t += 8) {
        int grp = t >> 4, sub = t & 15, mt = sub >> 2, nt = sub & 3;
        const __bf16* A = ihc + mt * 16 * 320;
        if (grp == 0)
            store_tile_bias(qf + mt * 16 * 64 + nt * 16, 64,
                            gemm_tile(A, 320, 64, Wq, nt), wq_b, nt * 16);
        else if (grp == 1)
            store_tile_bias(kf + mt * 16 * 64 + nt * 16, 64,
                            gemm_tile(A, 320, 64, Wk, nt), wk_b, nt * 16);
        else
            store_tile_bias(vf + mt * 16 * 64 + nt * 16, 64,
                            gemm_tile(A, 320, 320, Wv, nt), wv_b, nt * 16);
    }
    __syncthreads();

    // 8x8 masked softmax attention per agent-batch (tiny -> VALU)
    if (tid < 64) {
        const int r  = tid;
        const int i  = r & 7;
        const int rb = r - i;
        float logits[8], mx = -1e30f;
#pragma unroll
        for (int j = 0; j < 8; ++j) {
            if (j == i) logits[j] = -1e9f;
            else {
                float d = 0.f;
#pragma unroll 8
                for (int a = 0; a < 64; ++a)
                    d += qf[r * 64 + a] * kf[(rb + j) * 64 + a];
                logits[j] = d * 0.125f;          // 1/sqrt(A)
            }
            mx = fmaxf(mx, logits[j]);
        }
        float w[8], se = 0.f;
#pragma unroll
        for (int j = 0; j < 8; ++j) { w[j] = __expf(logits[j] - mx); se += w[j]; }
        float inv = 1.f / se;
        for (int a = 0; a < 64; ++a) {
            float acc = 0.f;
#pragma unroll
            for (int j = 0; j < 8; ++j) acc += w[j] * vf[(rb + j) * 64 + a];
            hc[r * 320 + 256 + a] = (__bf16)(acc * inv);
        }
    }
    __syncthreads();

    // fc2: [h|combined](K=320) @ Wfc2^T -> local_Q (8 tiles, one per wave)
    for (int t = wv; t < 8; t += 8) {
        int mt = t >> 1, nt = t & 1;
        v8f c = gemm_tile(hc + mt * 16 * 320, 320, 320, Wfc2, nt);
        const int lane = tid & 31, n = lane & 15, hi = lane >> 4;
        const float b = fc2_b[nt * 16 + n];
#pragma unroll
        for (int i = 0; i < 8; ++i) {
            int R = rowBase + mt * 16 + i + 8 * hi;
            localQ[(size_t)R * 32 + nt * 16 + n] = c[i] + b;
        }
    }
}

// ---------------------------------------------------------------------------
// Host launcher
// ---------------------------------------------------------------------------
extern "C" void kernel_launch(void* const* d_in, const int* in_sizes, int n_in,
                              void* d_out, int out_size, void* d_ws, size_t ws_size,
                              hipStream_t stream) {
    const float* inputs  = (const float*)d_in[0];
    const float* la      = (const float*)d_in[1];
    const float* hidden  = (const float*)d_in[2];
    const float* eps     = (const float*)d_in[3];
    const float* fc1_w   = (const float*)d_in[4];
    const float* fc1_b   = (const float*)d_in[5];
    const float* gw_ih   = (const float*)d_in[6];
    const float* gw_hh   = (const float*)d_in[7];
    const float* gb_ih   = (const float*)d_in[8];
    const float* gb_hh   = (const float*)d_in[9];
    const float* in1_w   = (const float*)d_in[10];
    const float* in1_b   = (const float*)d_in[11];
    const float* bn_g    = (const float*)d_in[12];
    const float* bn_b    = (const float*)d_in[13];
    const float* in2_w   = (const float*)d_in[14];
    const float* in2_b   = (const float*)d_in[15];
    const float* wq_w    = (const float*)d_in[16];
    const float* wq_b    = (const float*)d_in[17];
    const float* wk_w    = (const float*)d_in[18];
    const float* wk_b    = (const float*)d_in[19];
    const float* wv_w    = (const float*)d_in[20];
    const float* wv_b    = (const float*)d_in[21];
    const float* fc2_w   = (const float*)d_in[22];
    const float* fc2_b   = (const float*)d_in[23];

    // outputs: local_Q | h | intent | intent_embed (all f32)
    float* out       = (float*)d_out;
    float* o_localQ  = out;
    float* o_h       = out + (size_t)ROWS * NADIM;                       // 1048576
    float* o_intent  = o_h + (size_t)ROWS * HDIM;                        // +8388608
    float* o_iemb    = o_intent + (size_t)ROWS * IDIM;                   // +2097152

    // workspace carve-up (256B aligned)
    char* ws = (char*)d_ws;
    size_t off = 0;
    auto take = [&](size_t bytes) {
        void* p = ws + off;
        off += (bytes + 255) & ~(size_t)255;
        return p;
    };
    __bf16* Wfc1 = (__bf16*)take((size_t)256 * 256 * 2);
    __bf16* Wih  = (__bf16*)take((size_t)768 * 256 * 2);
    __bf16* Whh  = (__bf16*)take((size_t)768 * 256 * 2);
    __bf16* Win1 = (__bf16*)take((size_t)256 * 288 * 2);
    __bf16* Win2 = (__bf16*)take((size_t)128 * 256 * 2);
    __bf16* Wq   = (__bf16*)take((size_t)64 * 64 * 2);
    __bf16* Wk   = (__bf16*)take((size_t)64 * 64 * 2);
    __bf16* Wv   = (__bf16*)take((size_t)64 * 320 * 2);
    __bf16* Wfc2 = (__bf16*)take((size_t)32 * 320 * 2);
    float*  zbuf = (float*)take((size_t)ROWS * NHDIM * 4);
    float*  bn_sum   = (float*)take(256 * 4);
    float*  bn_sumsq = (float*)take(256 * 4);
    float*  bnS      = (float*)take(256 * 4);
    float*  bnB      = (float*)take(256 * 4);
    (void)ws_size; (void)n_in; (void)in_sizes; (void)out_size;

    // 0) zero BN accumulators (ws is poisoned, atomics need zeros every call)
    zero_kernel<<<2, 256, 0, stream>>>(bn_sum, 512);  // sum + sumsq contiguous? not guaranteed -> do both
    zero_kernel<<<1, 256, 0, stream>>>(bn_sumsq, 256);

    // 1) repack weights -> bf16 WMMA-B fragments (deterministic, every call)
    pack_b_kernel<<<(256 * 256) / 256, 256, 0, stream>>>(fc1_w, Wfc1, 256, 256);
    pack_b_kernel<<<(768 * 256) / 256, 256, 0, stream>>>(gw_ih, Wih, 768, 256);
    pack_b_kernel<<<(768 * 256) / 256, 256, 0, stream>>>(gw_hh, Whh, 768, 256);
    pack_b_kernel<<<(256 * 288) / 256, 256, 0, stream>>>(in1_w, Win1, 256, 288);
    pack_b_kernel<<<(128 * 256) / 256, 256, 0, stream>>>(in2_w, Win2, 128, 256);
    pack_b_kernel<<<(64 * 64) / 256, 256, 0, stream>>>(wq_w, Wq, 64, 64);
    pack_b_kernel<<<(64 * 64) / 256, 256, 0, stream>>>(wk_w, Wk, 64, 64);
    pack_b_kernel<<<(64 * 320) / 256, 256, 0, stream>>>(wv_w, Wv, 64, 320);
    pack_b_kernel<<<(32 * 320) / 256, 256, 0, stream>>>(fc2_w, Wfc2, 32, 320);

    // 2) fc1 + GRU + in1 (+BN partials)
    {
        size_t smemA = (size_t)(16 * 256) * 2 * 3      // inA, xbf, hbf
                     + (size_t)(16 * 288) * 2          // hla
                     + (size_t)(16 * 768) * 4 * 2      // gi, gh
                     + (size_t)(16 * 256) * 4;         // zt
        gru_in1_kernel<<<ROWS / 16, 256, smemA, stream>>>(
            inputs, la, hidden, fc1_b, gb_ih, gb_hh, in1_b,
            Wfc1, Wih, Whh, Win1, o_h, zbuf, bn_sum, bn_sumsq);
    }

    // 3) BN stats -> affine
    bn_finalize_kernel<<<1, 256, 0, stream>>>(bn_sum, bn_sumsq, bn_g, bn_b, bnS, bnB);

    // 4) intent + attention + fc2
    {
        size_t smemC = (size_t)(64 * 256) * 2          // act
                     + (size_t)(64 * 320) * 2 * 2      // ihc, hc
                     + (size_t)(64 * 128) * 4          // ipf
                     + (size_t)(64 * 64) * 4 * 3;      // qf, kf, vf
        attn_head_kernel<<<ROWS / 64, 256, smemC, stream>>>(
            zbuf, bnS, bnB, in2_b, eps, wq_b, wk_b, wv_b, fc2_b,
            Win2, Wq, Wk, Wv, Wfc2, o_h, o_localQ, o_intent, o_iemb);
    }
}
